// MaskRCNN_20435454394752
// MI455X (gfx1250) — matrix operand order repaired and checked by reading, not compiled, not run on hardware
//
#include <hip/hip_runtime.h>
#include <math.h>

#define NPROP 2000
#define NGT   100
#define PPOS  66
#define NNEG  134
#define MH    512
#define MW    512
#define CH    28
#define CW    28
#define RSTRIDE 520   // halfs per LDS row of R: 520*2B = 1040B -> 16B aligned, bank-conflict-free stride

typedef __attribute__((ext_vector_type(16))) _Float16 v16h;
typedef __attribute__((ext_vector_type(8)))  _Float16 v8h;
typedef __attribute__((ext_vector_type(8)))  float    v8f;

// ---------------------------------------------------------------------------
// Kernel 1: IoU matrix row-max + argmax per proposal (gt boxes staged in LDS)
// ---------------------------------------------------------------------------
__global__ void iou_argmax_kernel(const float* __restrict__ prop,
                                  const float* __restrict__ gt,
                                  float* __restrict__ iou_max,
                                  int* __restrict__ assign) {
    __shared__ float gtl[NGT * 4];
    for (int j = threadIdx.x; j < NGT * 4; j += blockDim.x) gtl[j] = gt[j];
    __syncthreads();

    int i = blockIdx.x * blockDim.x + threadIdx.x;
    if (i >= NPROP) return;

    float py1 = prop[i * 4 + 0], px1 = prop[i * 4 + 1];
    float py2 = prop[i * 4 + 2], px2 = prop[i * 4 + 3];
    float pa  = (py2 - py1) * (px2 - px1);

    float best = -INFINITY;
    int   bidx = 0;
    for (int g = 0; g < NGT; ++g) {
        float gy1 = gtl[g * 4 + 0], gx1 = gtl[g * 4 + 1];
        float gy2 = gtl[g * 4 + 2], gx2 = gtl[g * 4 + 3];
        float yy1 = fmaxf(py1, gy1), xx1 = fmaxf(px1, gx1);
        float yy2 = fminf(py2, gy2), xx2 = fminf(px2, gx2);
        float inter = fmaxf(xx2 - xx1, 0.f) * fmaxf(yy2 - yy1, 0.f);
        float ga    = (gy2 - gy1) * (gx2 - gx1);
        float uni   = pa + ga - inter;
        float iou   = (uni != 0.f) ? (inter / uni) : -1.f;
        if (iou > best) { best = iou; bidx = g; }   // first-max tie break == jnp.argmax
    }
    iou_max[i] = best;
    assign[i]  = bidx;
}

// ---------------------------------------------------------------------------
// Kernel 2: deterministic top-K by rank-select (matches lax.top_k ordering:
// descending score, ties -> lowest index). Masked scores staged in LDS.
// ---------------------------------------------------------------------------
__global__ void topk_rank_kernel(const float* __restrict__ scores,
                                 const float* __restrict__ iou_max,
                                 int K, int ge, int* __restrict__ out_idx) {
    __shared__ float s[NPROP];
    for (int j = threadIdx.x; j < NPROP; j += blockDim.x) {
        float im = iou_max[j];
        bool  c  = ge ? (im >= 0.5f) : (im < 0.5f);
        s[j] = c ? scores[j] : -INFINITY;
    }
    __syncthreads();

    int i = blockIdx.x * blockDim.x + threadIdx.x;
    if (i >= NPROP) return;
    float si   = s[i];
    int   rank = 0;
    for (int j = 0; j < NPROP; ++j) {
        float sj = s[j];
        rank += (sj > si) || ((sj == si) && (j < i));
    }
    if (rank < K) out_idx[rank] = i;   // ranks form a total order -> each slot written once
}

// ---------------------------------------------------------------------------
// Kernel 3: rois / class ids / deltas for all 200 slots; zero negative masks.
// ---------------------------------------------------------------------------
__global__ void finalize_kernel(const float* __restrict__ prop,
                                const float* __restrict__ gt,
                                const int*  __restrict__ gt_cls,
                                const int*  __restrict__ assign,
                                const int*  __restrict__ pos_idx,
                                const int*  __restrict__ neg_idx,
                                float* __restrict__ out) {
    float* rois   = out;            // [200,4]
    float* cls    = out + 800;      // [200]
    float* deltas = out + 1000;     // [200,4]
    float* masks  = out + 1800;     // [200,28,28]

    int b = blockIdx.x;             // 0..199
    if (b < PPOS) {
        int p = pos_idx[b];
        if (threadIdx.x == 0) {
            float py1 = prop[p * 4 + 0], px1 = prop[p * 4 + 1];
            float py2 = prop[p * 4 + 2], px2 = prop[p * 4 + 3];
            rois[b * 4 + 0] = py1; rois[b * 4 + 1] = px1;
            rois[b * 4 + 2] = py2; rois[b * 4 + 3] = px2;

            int g  = assign[p];
            cls[b] = (float)gt_cls[g];

            float gy1 = gt[g * 4 + 0], gx1 = gt[g * 4 + 1];
            float gy2 = gt[g * 4 + 2], gx2 = gt[g * 4 + 3];
            float h  = py2 - py1, w  = px2 - px1;
            float cy = py1 + 0.5f * h, cx = px1 + 0.5f * w;
            float gh = gy2 - gy1, gw = gx2 - gx1;
            float gcy = gy1 + 0.5f * gh, gcx = gx1 + 0.5f * gw;
            deltas[b * 4 + 0] = ((gcy - cy) / h) / 0.1f;
            deltas[b * 4 + 1] = ((gcx - cx) / w) / 0.1f;
            deltas[b * 4 + 2] = logf(gh / h) / 0.2f;
            deltas[b * 4 + 3] = logf(gw / w) / 0.2f;
        }
    } else {
        int j = b - PPOS;
        int p = neg_idx[j];
        if (threadIdx.x < 4) {
            rois[b * 4 + threadIdx.x]   = prop[p * 4 + threadIdx.x];
            deltas[b * 4 + threadIdx.x] = 0.f;
        }
        if (threadIdx.x == 0) cls[b] = 0.f;
        for (int t = threadIdx.x; t < CH * CW; t += blockDim.x)
            masks[(size_t)b * (CH * CW) + t] = 0.f;
    }
}

// ---------------------------------------------------------------------------
// Kernel 4: crop_and_resize via separable bilinear GEMM on WMMA.
//   R (28x512, f16 in LDS) = y-interpolated mask rows (coalesced row gathers)
//   Wx (512x28, 2 nnz/col)  = x-interp weights, generated per-lane in registers
//   out = R x Wx  -> 2x2 tiles of v_wmma_f32_16x16x32_f16, K=512 in 16 steps
// One block (4 waves) per positive ROI; wave w owns 16x16 output tile.
// ---------------------------------------------------------------------------
__global__ void mask_wmma_kernel(const float* __restrict__ prop,
                                 const float* __restrict__ gt_masks,
                                 const int*  __restrict__ pos_idx,
                                 const int*  __restrict__ assign,
                                 float* __restrict__ out_masks) {
    __shared__ __align__(16) _Float16 R[CH * RSTRIDE];   // 29.1 KB
    __shared__ int   y0s[CH], y1s[CH];
    __shared__ float wy0s[CH], wy1s[CH];
    __shared__ int   x0s[CW], x1s[CW];
    __shared__ float w0s[CW], w1s[CW];

    const int r = blockIdx.x;        // positive ROI slot 0..65
    const int p = pos_idx[r];
    const int g = assign[p];
    const float by1 = prop[p * 4 + 0], bx1 = prop[p * 4 + 1];
    const float by2 = prop[p * 4 + 2], bx2 = prop[p * 4 + 3];

    const int tid = threadIdx.x;

    // y-axis sampling params (threads 0..27)
    if (tid < CH) {
        float t   = (float)tid / (float)(CH - 1);
        float ys  = by1 * (MH - 1) + t * ((by2 - by1) * (MH - 1));
        float y0f = floorf(ys);
        float ly  = ys - y0f;
        bool  v   = (ys >= 0.f) && (ys <= (float)(MH - 1));
        y0s[tid]  = (int)fminf(fmaxf(y0f,       0.f), (float)(MH - 1));
        y1s[tid]  = (int)fminf(fmaxf(y0f + 1.f, 0.f), (float)(MH - 1));
        wy0s[tid] = v ? (1.f - ly) : 0.f;
        wy1s[tid] = v ? ly         : 0.f;
    }
    // x-axis sampling params (threads 32..59)
    if (tid >= 32 && tid < 32 + CW) {
        int   j   = tid - 32;
        float t   = (float)j / (float)(CW - 1);
        float xs  = bx1 * (MW - 1) + t * ((bx2 - bx1) * (MW - 1));
        float x0f = floorf(xs);
        float lx  = xs - x0f;
        bool  v   = (xs >= 0.f) && (xs <= (float)(MW - 1));
        x0s[j] = (int)fminf(fmaxf(x0f,       0.f), (float)(MW - 1));
        x1s[j] = (int)fminf(fmaxf(x0f + 1.f, 0.f), (float)(MW - 1));
        w0s[j] = v ? (1.f - lx) : 0.f;
        w1s[j] = v ? lx         : 0.f;
    }
    __syncthreads();

    // Build R: y-interpolated rows (full mask rows -> coalesced global loads)
    const float* mbase = gt_masks + (size_t)g * (MH * MW);
    for (int idx = tid; idx < CH * MW; idx += blockDim.x) {
        int i = idx >> 9;           // / 512
        int c = idx & (MW - 1);     // % 512
        float v0 = mbase[y0s[i] * MW + c];
        float v1 = mbase[y1s[i] * MW + c];
        R[i * RSTRIDE + c] = (_Float16)(wy0s[i] * v0 + wy1s[i] * v1);
    }
    __syncthreads();

    // WMMA: each wave computes one 16x16 tile of the 32x32-padded output
    const int wave = tid >> 5;
    const int lane = tid & 31;
    const int m0 = (wave & 1) * 16;
    const int n0 = (wave >> 1) * 16;
    const int ln = lane & 15;
    const int hi = (lane >= 16) ? 1 : 0;

    // B-column parameters for this lane's output column n
    const int n = n0 + ln;
    int   cx0 = -1, cx1 = -1;
    float cw0 = 0.f, cw1 = 0.f;
    if (n < CW) { cx0 = x0s[n]; cx1 = x1s[n]; cw0 = w0s[n]; cw1 = w1s[n]; }

    // A-row for this lane (lanes 0-15 and 16-31 share M, differ in K offset)
    const int  am    = m0 + ln;
    const bool am_ok = (am < CH);
    const int  aoff  = hi ? 8 : 0;   // halfs

    v8f acc = {};
    for (int k0 = 0; k0 < MW; k0 += 32) {
        // A fragment: two contiguous 8-half chunks per lane (16B DS loads)
        v16h a;
        if (am_ok) {
            const v8h alo = *(const v8h*)&R[am * RSTRIDE + k0 + aoff];
            const v8h ahi = *(const v8h*)&R[am * RSTRIDE + k0 + aoff + 16];
            #pragma unroll
            for (int e = 0; e < 8; ++e) { a[e] = alo[e]; a[8 + e] = ahi[e]; }
        } else {
            #pragma unroll
            for (int e = 0; e < 16; ++e) a[e] = (_Float16)0.f;
        }
        // B fragment: Wx[k, n] generated in registers (<=2 nnz per column)
        v16h b;
        const int kb = k0 + hi * 8;
        #pragma unroll
        for (int e = 0; e < 16; ++e) {
            int   k   = kb + e + ((e >> 3) << 3);
            float val = 0.f;
            if (k == cx0) val += cw0;
            if (k == cx1) val += cw1;   // x0==x1 (clamped) accumulates to 1.0
            b[e] = (_Float16)val;
        }
        acc = __builtin_amdgcn_wmma_f32_16x16x32_f16(
            /*neg_a=*/false, a, /*neg_b=*/false, b,
            /*c_mod=*/(short)0, acc, /*reuse_a=*/false, /*reuse_b=*/false);
    }

    // Writeback: C/D layout -> lane(0-15): M=v, lane(16-31): M=8+v ; N=lane&15
    #pragma unroll
    for (int v = 0; v < 8; ++v) {
        int m = m0 + v + hi * 8;
        if (m < CH && n < CW)
            out_masks[(size_t)r * (CH * CW) + m * CW + n] = rintf(acc[v]);  // RNE == jnp.round
    }
}

// ---------------------------------------------------------------------------
extern "C" void kernel_launch(void* const* d_in, const int* in_sizes, int n_in,
                              void* d_out, int out_size, void* d_ws, size_t ws_size,
                              hipStream_t stream) {
    const float* proposals  = (const float*)d_in[0];
    const float* gt_boxes   = (const float*)d_in[1];
    const int*   gt_cls     = (const int*)  d_in[2];
    const float* gt_masks   = (const float*)d_in[3];
    const float* pos_scores = (const float*)d_in[4];
    const float* neg_scores = (const float*)d_in[5];
    float* out = (float*)d_out;

    float* iou_max = (float*)d_ws;
    int*   assign  = (int*)((char*)d_ws + NPROP * 4);
    int*   pos_idx = (int*)((char*)d_ws + NPROP * 8);
    int*   neg_idx = (int*)((char*)d_ws + NPROP * 8 + PPOS * 4);

    const int blk = 256;
    const int ngr = (NPROP + blk - 1) / blk;

    iou_argmax_kernel<<<ngr, blk, 0, stream>>>(proposals, gt_boxes, iou_max, assign);
    topk_rank_kernel <<<ngr, blk, 0, stream>>>(pos_scores, iou_max, PPOS, 1, pos_idx);
    topk_rank_kernel <<<ngr, blk, 0, stream>>>(neg_scores, iou_max, NNEG, 0, neg_idx);
    finalize_kernel  <<<PPOS + NNEG, 64, 0, stream>>>(proposals, gt_boxes, gt_cls,
                                                      assign, pos_idx, neg_idx, out);
    mask_wmma_kernel <<<PPOS, 128, 0, stream>>>(proposals, gt_masks, pos_idx, assign,
                                                out + 1800);
}
